// DeepMemoryTransformer_16088947490840
// MI455X (gfx1250) — matrix operand order repaired and checked by reading, not compiled, not run on hardware
//
#include <hip/hip_runtime.h>

// ---------- CDNA5 WMMA / vector types ----------
typedef __attribute__((ext_vector_type(16))) __bf16         v16bf;
typedef __attribute__((ext_vector_type(8)))  float          v8f;
typedef __attribute__((ext_vector_type(16))) unsigned short v16us;
typedef __attribute__((ext_vector_type(4)))  unsigned int   v4u;
typedef __attribute__((ext_vector_type(8)))  int            v8i;
typedef __attribute__((ext_vector_type(4)))  int            v4i;

// Problem constants (from reference)
#define BB 2
#define SS 2048
#define QWW 1024
#define HH 16
#define MM 4096
#define DD 64
#define KK_TOP 32
#define OWW 1024
#define RR (BB*SS)          // 4096 query rows
#define HD (HH*DD)          // 1024

// fp32 -> bf16 round-to-nearest-even (bit pattern as ushort)
__device__ __forceinline__ unsigned short f2bf(float f){
    unsigned u = __builtin_bit_cast(unsigned, f);
    u += 0x7FFFu + ((u >> 16) & 1u);
    return (unsigned short)(u >> 16);
}

// A fragment: 16x32 bf16 tile from row-major A[row, kbase..kbase+31], ld elements.
// Per-lane: element e -> K = kbase + half*16 + e
// = one contiguous 16-element (32B) run -> two b128 loads.
struct V32B { v4u a, b; };
__device__ __forceinline__ v16bf load_a_frag(const unsigned short* __restrict__ A,
                                             int ld, int row, int kbase, int half){
    const unsigned short* base = A + (size_t)row * ld + kbase + half * 16;
    V32B t;
    t.a = *(const v4u*)(base);
    t.b = *(const v4u*)(base + 8);
    return __builtin_bit_cast(v16bf, t);
}

// B fragment from pre-packed layout: P points at the 32-element K-run for column n
// (packed [k/32][n][32], j = k%32). Lane needs j = half*16 + e, e=0..15:
// 16 contiguous ushorts -> two b128 loads (global or LDS).
__device__ __forceinline__ v16bf load_b_packed(const unsigned short* __restrict__ P, int half){
    return __builtin_bit_cast(v16bf, *(const v16us*)(P + half * 16));
}

__device__ __forceinline__ v8f wmma_bf16(v16bf a, v16bf b, v8f c){
    return __builtin_amdgcn_wmma_f32_16x16x32_bf16(false, a, false, b,
                                                   (short)0, c, false, false);
}

// ---------- fp32 -> bf16 elementwise ----------
__global__ void cvt_kernel(const float* __restrict__ in, unsigned short* __restrict__ out, int n){
    int i = blockIdx.x * blockDim.x + threadIdx.x;
    if (i < n) out[i] = f2bf(in[i]);
}

// ---------- fp32 -> bf16 with B-operand packing ----------
// in: batch of row-major [Kdim, N]; out: [batch][Kdim/32][N][32]
__global__ void pack_b_kernel(const float* __restrict__ W, unsigned short* __restrict__ out,
                              int Kdim, int N, int batch){
    int i = blockIdx.x * blockDim.x + threadIdx.x;
    int total = batch * Kdim * N;
    if (i >= total) return;
    int n = i % N;
    int k = (i / N) % Kdim;
    int b = i / (N * Kdim);
    out[((size_t)(b * (Kdim >> 5) + (k >> 5)) * N + n) * 32 + (k & 31)] = f2bf(W[i]);
}

// ---------- q = x @ Wq + bq -> bf16, layout [b*H+h][s][d] ----------
__global__ void qproj_kernel(const unsigned short* __restrict__ xb,
                             const unsigned short* __restrict__ Wqp,   // packed
                             const float* __restrict__ bq,
                             unsigned short* __restrict__ qb){
    int tid = threadIdx.x, wave = tid >> 5, lane = tid & 31;
    int t = blockIdx.x * 8 + wave;            // 16384 tiles = 256 x 64
    int tileM = t >> 6, tileN = t & 63;
    int half = lane >> 4, l16 = lane & 15;
    int row = tileM * 16 + l16;
    int col = tileN * 16 + l16;
    v8f c = {};
    for (int kk = 0; kk < QWW; kk += 32){
        if (kk + 32 < QWW)   // global_prefetch_b8 warm of next A panel
            __builtin_prefetch(xb + (size_t)row * QWW + kk + 32, 0, 3);
        v16bf a = load_a_frag(xb, QWW, row, kk, half);
        v16bf b = load_b_packed(Wqp + ((size_t)(kk >> 5) * HD + col) * 32, half);
        c = wmma_bf16(a, b, c);
    }
    int h = col >> 6, d = col & 63;
    float bias = bq[col];
#pragma unroll
    for (int r = 0; r < 8; ++r){
        int m = tileM * 16 + r + 8 * half;
        int b_ = m >> 11, s = m & (SS - 1);
        qb[(((size_t)(b_ * HH + h) * SS) + s) * DD + d] = f2bf(c[r] + bias);
    }
}

// ---------- keys = memory @ Wk + bk -> bf16, PACKED [h][d/32][m][32] ----------
__global__ void keys_kernel(const unsigned short* __restrict__ memb,
                            const unsigned short* __restrict__ Wkp,    // packed per head
                            const float* __restrict__ bk,
                            unsigned short* __restrict__ kpack){
    int tid = threadIdx.x, wave = tid >> 5, lane = tid & 31;
    int t = blockIdx.x * 8 + wave;            // 16384 tiles
    int h = t >> 10;
    int rem = t & 1023;
    int tileM = rem >> 2, tileN = rem & 3;
    int half = lane >> 4, l16 = lane & 15;
    const unsigned short* A  = memb + (size_t)h * MM * DD;
    const unsigned short* Bp = Wkp  + (size_t)h * (2 * DD * 32);
    int row = tileM * 16 + l16;               // m
    int col = tileN * 16 + l16;               // e (key dim)
    v8f c = {};
    for (int kk = 0; kk < DD; kk += 32){
        v16bf a = load_a_frag(A, DD, row, kk, half);
        v16bf b = load_b_packed(Bp + ((size_t)((kk >> 5) * DD + col)) * 32, half);
        c = wmma_bf16(a, b, c);
    }
    float bias = bk[h * DD + col];
#pragma unroll
    for (int r = 0; r < 8; ++r){
        int m = tileM * 16 + r + 8 * half;
        // packed B layout for score GEMM: [h][d>>5][m][d&31]
        kpack[((size_t)(h * 2 + (col >> 5)) * MM + m) * 32 + (col & 31)] = f2bf(c[r] + bias);
    }
}

// ---------- fused score GEMM + top-32 + sigmoid + gather + weighted sum ----------
// One workgroup (8 waves) per (b, h, 16-row s-tile). M in 16 chunks of 256 columns.
// Keys chunk is DMA'd by the Tensor Data Mover into LDS (tensor_load_to_lds),
// then consumed by WMMA via ds b128 fragment loads. Scores never leave LDS.
__global__ void score_topk_kernel(const unsigned short* __restrict__ qb,
                                  const unsigned short* __restrict__ kpack,
                                  const float* __restrict__ memory,
                                  unsigned short* __restrict__ preout){
    __shared__ unsigned short kt[2][256][32];   // keys chunk: [kw][m][k%32], 32 KB
    __shared__ float sc[16][256];               // chunk scores, 16 KB
    __shared__ float topv[16][KK_TOP];
    __shared__ int   topi[16][KK_TOP];
    __shared__ float gatev[16][KK_TOP];

    int tid = threadIdx.x, wave = tid >> 5, lane = tid & 31;
    int half = lane >> 4, l16 = lane & 15;
    int bh = blockIdx.x >> 7;               // b*16 + h
    int stile = blockIdx.x & 127;
    int h = bh & 15;
    int s0 = stile * 16;

    const unsigned short* qbase = qb + ((size_t)bh * SS + s0) * DD;

    // q tile (16x64): two A fragments per wave
    v16bf a0 = load_a_frag(qbase, DD, l16, 0, half);
    v16bf a1 = load_a_frag(qbase, DD, l16, 32, half);

    float rmin = -3.0e38f; int rpos = 0;
    if (tid < 16){
#pragma unroll
        for (int k = 0; k < KK_TOP; ++k){ topv[tid][k] = -3.0e38f; topi[tid][k] = 0; }
    }

    unsigned lds_off = (unsigned)(size_t)(void*)&kt[0][0][0];

    for (int ch = 0; ch < 16; ++ch){
        int mc = ch * 256;
        if (wave == 0){
            // ---- TDM: load keys chunk [2 kw-rows x 8192 elem] into LDS ----
            // Source: kpack + ((h*2+0)*MM + mc)*32 elements; kw-row stride = MM*32 elem.
            unsigned long long ga =
                (unsigned long long)(size_t)(const void*)(kpack + ((size_t)(h * 2) * MM + mc) * 32);
            v4u g0;
            g0[0] = 1u;                                  // count=1 (valid descriptor)
            g0[1] = lds_off;                             // lds_addr (bytes)
            g0[2] = (unsigned)(ga & 0xFFFFFFFFu);        // global_addr[31:0]
            g0[3] = (unsigned)((ga >> 32) & 0x1FFFFFFu)  // global_addr[56:32]
                    | (2u << 30);                        // type=2 ("image")
            unsigned td0 = 8192u, td1 = 2u, tile0 = 8192u, tile1 = 2u;
            unsigned long long str0 = (unsigned long long)MM * 32ull;  // 131072 elements
            v8i g1;
            g1[0] = (int)(1u << 16);                     // data_size=1 (2 bytes)
            g1[1] = (int)((td0 & 0xFFFFu) << 16);        // tensor_dim0[15:0]
            g1[2] = (int)(((td0 >> 16) & 0xFFFFu) | ((td1 & 0xFFFFu) << 16));
            g1[3] = (int)(((td1 >> 16) & 0xFFFFu) | (tile0 << 16));   // tile_dim0
            g1[4] = (int)(tile1);                        // tile_dim1; tile_dim2=0
            g1[5] = (int)(unsigned)(str0 & 0xFFFFFFFFu); // tensor_dim0_stride[31:0]
            g1[6] = (int)(unsigned)((str0 >> 32) & 0xFFFFu);
            g1[7] = 0;
            v4i z4 = {0, 0, 0, 0};
            v8i z8 = {0, 0, 0, 0, 0, 0, 0, 0};
            // 6-arg form on this toolchain: (g0, g1, g2, g3, g4, cpol)
            __builtin_amdgcn_tensor_load_to_lds(g0, g1, z4, z4, z8, 0);
            __builtin_amdgcn_s_wait_tensorcnt(0);
        }
        __syncthreads();    // kt ready for everyone

        // ---- score WMMAs: each wave owns 32 columns (2 N-tiles) ----
#pragma unroll
        for (int nt = 0; nt < 2; ++nt){
            int mlocal = wave * 32 + nt * 16;
            v16bf b0 = load_b_packed(&kt[0][mlocal + l16][0], half);
            v16bf b1 = load_b_packed(&kt[1][mlocal + l16][0], half);
            v8f c = {};
            c = wmma_bf16(a0, b0, c);
            c = wmma_bf16(a1, b1, c);
#pragma unroll
            for (int r = 0; r < 8; ++r)
                sc[r + 8 * half][mlocal + l16] = c[r];
        }
        __syncthreads();    // sc ready; kt reads done

        // ---- running top-K merge, one thread per query row (order-invariant set) ----
        if (tid < 16){
            for (int j = 0; j < 256; ++j){
                float v = sc[tid][j];
                if (v > rmin){
                    topv[tid][rpos] = v; topi[tid][rpos] = mc + j;
                    rmin = topv[tid][0]; rpos = 0;
#pragma unroll
                    for (int k = 1; k < KK_TOP; ++k){
                        float tv = topv[tid][k];
                        if (tv < rmin){ rmin = tv; rpos = k; }
                    }
                }
            }
        }
        __syncthreads();    // selection done; safe to overwrite kt/sc
    }

    if (tid < 16){
#pragma unroll
        for (int k = 0; k < KK_TOP; ++k)
            gatev[tid][k] = 1.0f / (1.0f + expf(-topv[tid][k]));
    }
    __syncthreads();

    // ---- gated weighted sum of gathered memory rows (fp32, L2-resident) ----
    int row = tid >> 4;
    int d0 = (tid & 15) * 4;
    float acc0 = 0.f, acc1 = 0.f, acc2 = 0.f, acc3 = 0.f;
    const float* mbase = memory + (size_t)h * MM * DD;
    for (int k = 0; k < KK_TOP; ++k){
        float g = gatev[row][k];
        int idx = topi[row][k];
        const float4 mv = *(const float4*)(mbase + (size_t)idx * DD + d0);
        acc0 += g * mv.x; acc1 += g * mv.y; acc2 += g * mv.z; acc3 += g * mv.w;
    }
    int b_ = bh >> 4;
    size_t obase = ((size_t)(b_ * SS + s0 + row)) * HD + h * DD + d0;
    preout[obase + 0] = f2bf(acc0);
    preout[obase + 1] = f2bf(acc1);
    preout[obase + 2] = f2bf(acc2);
    preout[obase + 3] = f2bf(acc3);
}

// ---------- out = preout @ Wo + bo -> fp32 ----------
__global__ void outproj_kernel(const unsigned short* __restrict__ pre,
                               const unsigned short* __restrict__ Wop,  // packed
                               const float* __restrict__ bo,
                               float* __restrict__ out){
    int tid = threadIdx.x, wave = tid >> 5, lane = tid & 31;
    int t = blockIdx.x * 8 + wave;
    int tileM = t >> 6, tileN = t & 63;
    int half = lane >> 4, l16 = lane & 15;
    int row = tileM * 16 + l16;
    int col = tileN * 16 + l16;
    v8f c = {};
    for (int kk = 0; kk < HD; kk += 32){
        v16bf a = load_a_frag(pre, HD, row, kk, half);
        v16bf b = load_b_packed(Wop + ((size_t)(kk >> 5) * OWW + col) * 32, half);
        c = wmma_bf16(a, b, c);
    }
    float bias = bo[col];
#pragma unroll
    for (int r = 0; r < 8; ++r){
        int m = tileM * 16 + r + 8 * half;
        out[(size_t)m * OWW + col] = c[r] + bias;
    }
}

extern "C" void kernel_launch(void* const* d_in, const int* in_sizes, int n_in,
                              void* d_out, int out_size, void* d_ws, size_t ws_size,
                              hipStream_t stream){
    (void)in_sizes; (void)n_in; (void)out_size; (void)ws_size;
    const float* x   = (const float*)d_in[0];
    const float* mem = (const float*)d_in[1];
    const float* Wq  = (const float*)d_in[2];
    const float* bq  = (const float*)d_in[3];
    const float* Wk  = (const float*)d_in[4];
    const float* bk  = (const float*)d_in[5];
    const float* Wo  = (const float*)d_in[6];
    const float* bo  = (const float*)d_in[7];
    float* out = (float*)d_out;

    // workspace layout (bf16 as ushort), 256B aligned chunks, ~44 MiB total
    char* w = (char*)d_ws;
    auto take = [&](size_t elems)->unsigned short*{
        unsigned short* p = (unsigned short*)w;
        w += ((elems * 2 + 255) / 256) * 256;
        return p;
    };
    unsigned short* xb    = take((size_t)RR * QWW);
    unsigned short* memb  = take((size_t)HH * MM * DD);
    unsigned short* Wqp   = take((size_t)QWW * HD);        // packed [K/32][N][32]
    unsigned short* Wkp   = take((size_t)HH * 2 * DD * 32);
    unsigned short* Wop   = take((size_t)HD * OWW);
    unsigned short* qb    = take((size_t)RR * HD);
    unsigned short* kpack = take((size_t)HH * 2 * MM * 32);
    unsigned short* pre   = take((size_t)RR * HD);

    cvt_kernel<<<(RR * QWW + 255) / 256, 256, 0, stream>>>(x, xb, RR * QWW);
    cvt_kernel<<<(HH * MM * DD + 255) / 256, 256, 0, stream>>>(mem, memb, HH * MM * DD);
    pack_b_kernel<<<(QWW * HD + 255) / 256, 256, 0, stream>>>(Wq, Wqp, QWW, HD, 1);
    pack_b_kernel<<<(HH * DD * DD + 255) / 256, 256, 0, stream>>>(Wk, Wkp, DD, DD, HH);
    pack_b_kernel<<<(HD * OWW + 255) / 256, 256, 0, stream>>>(Wo, Wop, HD, OWW, 1);

    qproj_kernel     <<<2048, 256, 0, stream>>>(xb, Wqp, bq, qb);
    keys_kernel      <<<2048, 256, 0, stream>>>(memb, Wkp, bk, kpack);
    score_topk_kernel<<<4096, 256, 0, stream>>>(qb, kpack, mem, pre);
    outproj_kernel   <<<2048, 256, 0, stream>>>(pre, Wop, bo, out);
}
